// ProtectedChannelResidualCodebookReadout_84928683311889
// MI455X (gfx1250) — matrix operand order repaired and checked by reading, not compiled, not run on hardware
//
#include <hip/hip_runtime.h>
#include <hip/hip_bf16.h>
#include <math.h>

typedef __bf16 bf16_t;
typedef __attribute__((ext_vector_type(4)))  __bf16 v4bf;
typedef __attribute__((ext_vector_type(8)))  __bf16 v8bf;
typedef __attribute__((ext_vector_type(16))) __bf16 v16bf;
typedef __attribute__((ext_vector_type(2)))  float  v2f;
typedef __attribute__((ext_vector_type(4)))  float  v4f;
typedef __attribute__((ext_vector_type(8)))  float  v8f;

#define INVSCALE 0.044194173824159216f  /* 1/sqrt(512) */

// ---------------------------------------------------------------------------
// Kernel 0: codebook -> bf16 (row major, B operand for GEMM1) and
//           cbT[p][c] = codebook[c][p] for p<64 (B operand for slot_base GEMM)
// ---------------------------------------------------------------------------
__global__ __launch_bounds__(256) void prep_cb(const float* __restrict__ cb,
                                               bf16_t* __restrict__ cb16,
                                               bf16_t* __restrict__ cbT) {
    int c = blockIdx.x;          // 0..511
    int t = threadIdx.x;         // 0..255
    cb16[(size_t)c * 512 + t]       = (bf16_t)cb[(size_t)c * 512 + t];
    cb16[(size_t)c * 512 + t + 256] = (bf16_t)cb[(size_t)c * 512 + t + 256];
    if (t < 64) cbT[(size_t)t * 512 + c] = (bf16_t)cb[(size_t)c * 512 + t];
}

// ---------------------------------------------------------------------------
// Kernel 1: q-side: q_proj, code_w softmax, query_base, prot_code_w softmax
// one block per n, 256 threads
// ---------------------------------------------------------------------------
__global__ __launch_bounds__(256) void qside(const float* __restrict__ q,
                                             const float* __restrict__ Wq,
                                             const float* __restrict__ cb,
                                             const float* __restrict__ pcb,
                                             float* __restrict__ code_w,
                                             float* __restrict__ prot_code_w) {
    __shared__ float qp[512];
    __shared__ float cw[512];
    __shared__ float red[256];
    __shared__ float pres[64];
    __shared__ float s01[2];
    int n = blockIdx.x, t = threadIdx.x;
    const float* qn = q + (size_t)n * 512;

    // q_proj[d] = sum_k q[n,k] * Wq[d,k]
    for (int rep = 0; rep < 2; ++rep) {
        int d = t + rep * 256;
        const v4f* wr = (const v4f*)(Wq + (size_t)d * 512);
        const v4f* q4 = (const v4f*)qn;
        float acc = 0.f;
        for (int k = 0; k < 128; ++k) {
            v4f a = q4[k], b = wr[k];
            acc += a.x * b.x + a.y * b.y + a.z * b.z + a.w * b.w;
        }
        qp[d] = acc;
    }
    __syncthreads();
    // logits over codebook
    for (int rep = 0; rep < 2; ++rep) {
        int c = t + rep * 256;
        const v4f* cr = (const v4f*)(cb + (size_t)c * 512);
        const v4f* q4 = (const v4f*)qp;
        float acc = 0.f;
        for (int k = 0; k < 128; ++k) {
            v4f a = q4[k], b = cr[k];
            acc += a.x * b.x + a.y * b.y + a.z * b.z + a.w * b.w;
        }
        cw[c] = acc * INVSCALE;
    }
    __syncthreads();
    // softmax over 512
    float m = fmaxf(cw[t], cw[t + 256]);
    red[t] = m; __syncthreads();
    for (int off = 128; off; off >>= 1) {
        if (t < off) red[t] = fmaxf(red[t], red[t + off]);
        __syncthreads();
    }
    float mx = red[0]; __syncthreads();
    float e0 = expf(cw[t] - mx), e1 = expf(cw[t + 256] - mx);
    red[t] = e0 + e1; __syncthreads();
    for (int off = 128; off; off >>= 1) {
        if (t < off) red[t] += red[t + off];
        __syncthreads();
    }
    float inv = 1.0f / red[0]; __syncthreads();
    cw[t] = e0 * inv;
    cw[t + 256] = e1 * inv;
    __syncthreads();
    code_w[(size_t)n * 512 + t]       = cw[t];
    code_w[(size_t)n * 512 + t + 256] = cw[t + 256];
    // prot residual of query: (q_proj - code_w@codebook)[:64]
    if (t < 64) {
        float qb = 0.f;
        for (int c = 0; c < 512; ++c) qb += cw[c] * cb[(size_t)c * 512 + t];
        pres[t] = qp[t] - qb;
    }
    __syncthreads();
    if (t < 64) {
        const float* pr = pcb + t * 64;
        float acc = 0.f;
        for (int p = 0; p < 64; ++p) acc += pres[p] * pr[p];
        red[t] = acc * INVSCALE;
    }
    __syncthreads();
    if (t == 0) {
        float mx2 = -1e30f;
        for (int r = 0; r < 64; ++r) mx2 = fmaxf(mx2, red[r]);
        float s = 0.f;
        for (int r = 0; r < 64; ++r) s += expf(red[r] - mx2);
        s01[0] = mx2; s01[1] = 1.0f / s;
    }
    __syncthreads();
    if (t < 64) prot_code_w[(size_t)n * 64 + t] = expf(red[t] - s01[0]) * s01[1];
}

// ---------------------------------------------------------------------------
// Kernel 2: heavy per (n, s-tile of 32) WMMA kernel.
// 128 threads = 4 waves. wave w: rows 16*(w&1), c-tiles split by (w>>1).
// Softmax phases parallelized 4 threads/row.
// ---------------------------------------------------------------------------
__global__ __launch_bounds__(128) void slotw(const float*  __restrict__ K,
                                             const bf16_t* __restrict__ cb16,
                                             const bf16_t* __restrict__ cbT,
                                             const float*  __restrict__ pcb,
                                             const float*  __restrict__ code_w,
                                             const float*  __restrict__ prot_code_w,
                                             const float*  __restrict__ prot_logit,
                                             float* __restrict__ wsum,
                                             float* __restrict__ eff_out) {
    constexpr int ROWS = 32;
    __shared__ __align__(32) bf16_t Kb[ROWS][512];   // K tile (bf16); reused as slot weights
    __shared__ __align__(32) float  Lg[ROWS][512];   // logits; later carved into PC / LR
    __shared__ float SB[ROWS][64];
    __shared__ float bwv[ROWS];
    __shared__ float red4a[ROWS][4];
    __shared__ float red4b[ROWS][4];
    __shared__ float rowmax[ROWS];
    __shared__ float rowinv[ROWS];

    int n  = blockIdx.x >> 4;              // S/32 = 16 tiles per n
    int s0 = (blockIdx.x & 15) * ROWS;
    int t = threadIdx.x;
    int lane = t & 31, wave = t >> 5;
    const float* Kn = K + ((size_t)n * 512 + s0) * 512;

    // phase 1: K tile -> bf16 LDS (tile is one contiguous 32*512-float block)
    {
        const v4f* Kn4 = (const v4f*)Kn;
        bf16_t* Kbf = &Kb[0][0];
        for (int i = t; i < ROWS * 512 / 4; i += 128) {
            v4f x = Kn4[i];
            v4bf y;
            y[0] = (bf16_t)x.x; y[1] = (bf16_t)x.y;
            y[2] = (bf16_t)x.z; y[3] = (bf16_t)x.w;
            *(v4bf*)(Kbf + i * 4) = y;
        }
    }
    __syncthreads();

    // phase 2: logits[s, c] = (Kb @ codebook^T) * invscale via WMMA bf16
    {
        int rbase = 16 * (wave & 1);
        int arow  = rbase + (lane & 15);
        int aoff  = (lane >> 4) << 4;     // +16 elements for upper half-wave (A layout)
        int boff  = (lane >> 4) << 4;     // +16 elements for upper half-wave (B layout)
        int ct0   = (wave >> 1) * 16;
        for (int ct = ct0; ct < ct0 + 16; ++ct) {
            v8f acc = {};
            int crow = ct * 16 + (lane & 15);
            for (int k = 0; k < 512; k += 32) {
                v8bf alo = *(const v8bf*)&Kb[arow][k + aoff];
                v8bf ahi = *(const v8bf*)&Kb[arow][k + aoff + 8];
                v16bf a = __builtin_shufflevector(alo, ahi,
                          0,1,2,3,4,5,6,7,8,9,10,11,12,13,14,15);
                v16bf b = *(const v16bf*)&cb16[(size_t)crow * 512 + k + boff];
                acc = __builtin_amdgcn_wmma_f32_16x16x32_bf16(
                          false, a, false, b, (short)0, acc, false, false);
            }
            int nn = lane & 15, mb = (lane >> 4) << 3;
            #pragma unroll
            for (int j = 0; j < 8; ++j)
                Lg[rbase + mb + j][ct * 16 + nn] = acc[j] * INVSCALE;
        }
    }
    __syncthreads();

    // phase 3: row softmax + base_weights dot (4 threads per row, 128 cols each)
    const float* cwn = code_w + (size_t)n * 512;
    {
        int row = t >> 2, g = t & 3;
        int c0 = g * 128;
        float mx = -1e30f;
        for (int c = c0; c < c0 + 128; ++c) mx = fmaxf(mx, Lg[row][c]);
        red4a[row][g] = mx;
        __syncthreads();
        if (g == 0)
            rowmax[row] = fmaxf(fmaxf(red4a[row][0], red4a[row][1]),
                                fmaxf(red4a[row][2], red4a[row][3]));
        __syncthreads();
        float rmx = rowmax[row];
        float s = 0.f, bwacc = 0.f;
        for (int c = c0; c < c0 + 128; ++c) {
            float e = expf(Lg[row][c] - rmx);
            s += e;
            bwacc += e * cwn[c];
        }
        red4a[row][g] = s;
        red4b[row][g] = bwacc;
        __syncthreads();
        if (g == 0) {
            float st = red4a[row][0] + red4a[row][1] + red4a[row][2] + red4a[row][3];
            float bt = red4b[row][0] + red4b[row][1] + red4b[row][2] + red4b[row][3];
            float inv = 1.0f / st;
            bwv[row] = bt * inv;
            rowinv[row] = inv;
        }
        __syncthreads();
        float rinv = rowinv[row];
        for (int c = c0; c < c0 + 128; ++c)
            Kb[row][c] = (bf16_t)(expf(Lg[row][c] - rmx) * rinv);
    }
    __syncthreads();

    // phase 4: slot_base[s, p] = weights @ codebook[:, :64]  (contraction over c)
    {
        int rbase = 16 * (wave & 1);
        int arow  = rbase + (lane & 15);
        int aoff  = (lane >> 4) << 4;
        int boff  = (lane >> 4) << 4;
        for (int pi = 0; pi < 2; ++pi) {
            int pt = (wave >> 1) + pi * 2;
            v8f acc = {};
            int prow = pt * 16 + (lane & 15);
            for (int k = 0; k < 512; k += 32) {
                v8bf alo = *(const v8bf*)&Kb[arow][k + aoff];
                v8bf ahi = *(const v8bf*)&Kb[arow][k + aoff + 8];
                v16bf a = __builtin_shufflevector(alo, ahi,
                          0,1,2,3,4,5,6,7,8,9,10,11,12,13,14,15);
                v16bf b = *(const v16bf*)&cbT[(size_t)prow * 512 + k + boff];
                acc = __builtin_amdgcn_wmma_f32_16x16x32_bf16(
                          false, a, false, b, (short)0, acc, false, false);
            }
            int nn = lane & 15, mb = (lane >> 4) << 3;
            #pragma unroll
            for (int j = 0; j < 8; ++j)
                SB[rbase + mb + j][pt * 16 + nn] = acc[j];
        }
    }
    // stage prot_codebook into freed Lg space
    float* PC = &Lg[0][0];          // 64*64 floats
    float* LR = &Lg[0][0] + 4096;   // 32*64 floats
    for (int i = t; i < 64 * 64; i += 128) PC[i] = pcb[i];
    __syncthreads();

    // phase 5: prot residual softmax over R=64 (4 threads per row, 16 r each)
    float gate = 1.0f / (1.0f + expf(-prot_logit[0]));
    {
        int row = t >> 2, g = t & 3;
        float res[64];
        const float* Krow = Kn + (size_t)row * 512;
        #pragma unroll
        for (int p = 0; p < 64; ++p) res[p] = Krow[p] - SB[row][p];
        float mx = -1e30f;
        for (int r = g * 16; r < g * 16 + 16; ++r) {
            float acc = 0.f;
            #pragma unroll
            for (int p = 0; p < 64; ++p) acc += res[p] * PC[r * 64 + p];
            acc *= INVSCALE;
            LR[row * 64 + r] = acc;
            mx = fmaxf(mx, acc);
        }
        red4a[row][g] = mx;
        __syncthreads();
        if (g == 0)
            rowmax[row] = fmaxf(fmaxf(red4a[row][0], red4a[row][1]),
                                fmaxf(red4a[row][2], red4a[row][3]));
        __syncthreads();
        float rmx = rowmax[row];
        const float* pcwn = prot_code_w + (size_t)n * 64;
        float s = 0.f, pacc = 0.f;
        for (int r = g * 16; r < g * 16 + 16; ++r) {
            float e = expf(LR[row * 64 + r] - rmx);
            s += e;
            pacc += e * pcwn[r];
        }
        red4a[row][g] = s;
        red4b[row][g] = pacc;
        __syncthreads();
        if (g == 0) {
            float st = red4a[row][0] + red4a[row][1] + red4a[row][2] + red4a[row][3];
            float pt_ = red4b[row][0] + red4b[row][1] + red4b[row][2] + red4b[row][3];
            float prw = pt_ / st;
            float wc = bwv[row] + gate * prw;
            wsum[(size_t)n * 512 + s0 + row]    = wc;
            eff_out[(size_t)n * 512 + s0 + row] = wc / (1.0f + gate);
        }
    }
}

// ---------------------------------------------------------------------------
// Kernel 3: summary[n,d] = sum_s wsum[n,s] * V[n,s,d]   (bandwidth: streams V)
// ---------------------------------------------------------------------------
__global__ __launch_bounds__(256) void summarize(const float* __restrict__ V,
                                                 const float* __restrict__ wsum,
                                                 float* __restrict__ summ) {
    __shared__ float w[512];
    int n = blockIdx.x, t = threadIdx.x;
    for (int i = t; i < 512; i += 256) w[i] = wsum[(size_t)n * 512 + i];
    __syncthreads();
    const float* Vn = V + (size_t)n * 512 * 512;
    int d0 = t * 2;
    float a0 = 0.f, a1 = 0.f;
    for (int s = 0; s < 512; ++s) {
        v2f v = *(const v2f*)&Vn[(size_t)s * 512 + d0];
        float ws_ = w[s];
        a0 += ws_ * v.x;
        a1 += ws_ * v.y;
    }
    *(v2f*)&summ[(size_t)n * 512 + d0] = (v2f){a0, a1};
}

// ---------------------------------------------------------------------------
// Kernel 4: logits = summary @ clf_W^T + clf_b ; recon = summary @ rec_W^T + rec_b
// ---------------------------------------------------------------------------
__global__ __launch_bounds__(256) void heads(const float* __restrict__ summ,
                                             const float* __restrict__ clfW,
                                             const float* __restrict__ clfb,
                                             const float* __restrict__ recW,
                                             const float* __restrict__ recb,
                                             float* __restrict__ out) {
    int id = blockIdx.x * 256 + threadIdx.x;
    if (id < 256 * 1000) {
        int n = id / 1000, c = id % 1000;
        const v4f* sn = (const v4f*)(summ + (size_t)n * 512);
        const v4f* wr = (const v4f*)(clfW + (size_t)c * 512);
        float acc = clfb[c];
        for (int d = 0; d < 128; ++d) {
            v4f a = sn[d], b = wr[d];
            acc += a.x * b.x + a.y * b.y + a.z * b.z + a.w * b.w;
        }
        out[id] = acc;
    } else if (id < 256 * 1000 + 256 * 512) {
        int id2 = id - 256000;
        int n = id2 / 512, dd = id2 % 512;
        const v4f* sn = (const v4f*)(summ + (size_t)n * 512);
        const v4f* wr = (const v4f*)(recW + (size_t)dd * 512);
        float acc = recb[dd];
        for (int d = 0; d < 128; ++d) {
            v4f a = sn[d], b = wr[d];
            acc += a.x * b.x + a.y * b.y + a.z * b.z + a.w * b.w;
        }
        out[256000 + id2] = acc;
    }
}

// ---------------------------------------------------------------------------
extern "C" void kernel_launch(void* const* d_in, const int* in_sizes, int n_in,
                              void* d_out, int out_size, void* d_ws, size_t ws_size,
                              hipStream_t stream) {
    const float* q    = (const float*)d_in[0];
    const float* K    = (const float*)d_in[1];
    const float* V    = (const float*)d_in[2];
    const float* cb   = (const float*)d_in[3];
    const float* pcb  = (const float*)d_in[4];
    const float* Wq   = (const float*)d_in[5];
    const float* plg  = (const float*)d_in[6];
    const float* clfW = (const float*)d_in[7];
    const float* clfb = (const float*)d_in[8];
    const float* recW = (const float*)d_in[9];
    const float* recb = (const float*)d_in[10];

    char* ws = (char*)d_ws;
    bf16_t* cb16   = (bf16_t*)(ws + 0);        // 512*512*2  = 524288
    bf16_t* cbT    = (bf16_t*)(ws + 524288);   // 64*512*2   = 65536
    float*  code_w = (float*)(ws + 589824);    // 256*512*4  = 524288
    float*  pcw    = (float*)(ws + 1114112);   // 256*64*4   = 65536
    float*  wsum   = (float*)(ws + 1179648);   // 256*512*4  = 524288
    float*  summ   = (float*)(ws + 1703936);   // 256*512*4  = 524288

    float* out = (float*)d_out;
    // out layout: logits [256*1000] | recon [256*512] | eff_weights [256*512]
    float* eff_out = out + 256 * 1000 + 256 * 512;

    prep_cb<<<512, 256, 0, stream>>>(cb, cb16, cbT);
    qside<<<256, 256, 0, stream>>>(q, Wq, cb, pcb, code_w, pcw);
    slotw<<<256 * 16, 128, 0, stream>>>(K, cb16, cbT, pcb, code_w, pcw, plg,
                                        wsum, eff_out);
    summarize<<<256, 256, 0, stream>>>(V, wsum, summ);
    heads<<<1512, 256, 0, stream>>>(summ, clfW, clfb, recW, recb, out);
}
